// SelfInteraction_49589692399713
// MI455X (gfx1250) — matrix operand order repaired and checked by reading, not compiled, not run on hardware
//
#include <hip/hip_runtime.h>
#include <hip/hip_bf16.h>

typedef __attribute__((ext_vector_type(16))) _Float16 v16h;
typedef __attribute__((ext_vector_type(8)))  _Float16 v8h;
typedef __attribute__((ext_vector_type(8)))  float    v8f;

union V16 { v16h v; v8h h[2]; };

#define NBATCH   50000
#define NTILES   (NBATCH / 16)          // 3125
#define WAVES    8

// scales baked into the f16 weight panels
#define SCALE000 0.0139754249f          // c_out0 = sqrt(1/(64*64+32*32))
#define SCALE110 0.0080687151f          // c_out0 / sqrt(3)
#define SCALEV   0.015625f              // c_out1 / sqrt(3) = 1/64

// -----------------------------------------------------------------------------
// Weight reorder kernels: pack f32 weights into f16 B-matrix panels laid out in
// exactly the per-WMMA VGPR order:  [n_tile][k_chunk][lane][16 halves], where
// for lane L, half h:  K = k_chunk*32 + (L>>4)*8 + (h&7) + (h>>3)*16,  N = n_tile*16 + (L&15).
// Note the flat feature index k equals (u*dim_v + v) by construction, so the
// source offset is simply K*Wcols + N.
// -----------------------------------------------------------------------------
__global__ void prep_B000(const float* __restrict__ W, _Float16* __restrict__ B) {
  int id = blockIdx.x * 256 + threadIdx.x;                 // 262144 = 4*128*512
  int h = id & 15, lane = (id >> 4) & 31, c = (id >> 9) & 127, t = id >> 16;
  int K = c * 32 + ((lane >> 4) << 3) + (h & 7) + ((h >> 3) << 4); // u*64+v, u=K>>6
  int N = t * 16 + (lane & 15);
  B[id] = (_Float16)(SCALE000 * W[K * 64 + N]);
}

__global__ void prep_B110(const float* __restrict__ W, _Float16* __restrict__ B) {
  int id = blockIdx.x * 256 + threadIdx.x;                 // 65536 = 4*32*512
  int h = id & 15, lane = (id >> 4) & 31, c = (id >> 9) & 31, t = id >> 14;
  int K = c * 32 + ((lane >> 4) << 3) + (h & 7) + ((h >> 3) << 4); // u*32+v
  int N = t * 16 + (lane & 15);
  B[id] = (_Float16)(SCALE110 * W[K * 64 + N]);
}

__global__ void prep_BV(const float* __restrict__ W011,
                        const float* __restrict__ W101,
                        _Float16* __restrict__ B) {
  int id = blockIdx.x * 256 + threadIdx.x;                 // 131072 = 2*128*512
  int h = id & 15, lane = (id >> 4) & 31, c = (id >> 9) & 127, t = id >> 16;
  int K = c * 32 + ((lane >> 4) << 3) + (h & 7) + ((h >> 3) << 4); // 0..4095
  int N = t * 16 + (lane & 15);
  float w = (K < 2048) ? W011[K * 32 + N]                  // K = u*32+v (s ⊗ v)
                       : W101[(K - 2048) * 32 + N];        // K-2048 = u*64+v (v ⊗ s)
  B[id] = (_Float16)(SCALEV * w);
}

// -----------------------------------------------------------------------------
// Main kernel: 8 waves/block, one 16-row batch tile per wave.
// -----------------------------------------------------------------------------
__global__ __launch_bounds__(256) void si_main(
    const float* __restrict__ x,
    const _Float16* __restrict__ B000,
    const _Float16* __restrict__ B110,
    const _Float16* __restrict__ BV,
    float* __restrict__ out, int ntiles) {

  __shared__ _Float16 sS[WAVES][16][64];       // scalars, f16
  __shared__ _Float16 sV[WAVES][3][16][32];    // vector comps, f16, [i][row][u]

  const int lane = threadIdx.x & 31;
  const int wave = threadIdx.x >> 5;
  const int tile = blockIdx.x * WAVES + wave;
  if (tile >= ntiles) return;                  // wave-uniform exit (EXEC stays all-1)
  const int b0 = tile * 16;

  // ---- stage x tile -> LDS as f16 (wave-local, DScnt ordering is per-wave) ----
  const float* __restrict__ xin = x + (size_t)b0 * 160;
  for (int idx = lane; idx < 16 * 160; idx += 32) {
    int r = idx / 160;
    int c = idx - r * 160;
    float val = xin[idx];
    if (c < 64) {
      sS[wave][r][c] = (_Float16)val;
    } else {
      int cc = c - 64;
      int vv = cc / 3;
      int ii = cc - vv * 3;
      sV[wave][ii][r][vv] = (_Float16)val;
    }
  }

  const int row = lane & 15;
  const int og  = lane >> 4;   // K-group: lanes 0-15 take offsets {0..7,16..23}, 16-31 take {8..15,24..31}

  const v8h* __restrict__ srow  = (const v8h*)(&sS[wave][row][0]);     // 8 groups of 8 halves
  const v8h* __restrict__ vrow0 = (const v8h*)(&sV[wave][0][row][0]);  // 4 groups
  const v8h* __restrict__ vrow1 = (const v8h*)(&sV[wave][1][row][0]);
  const v8h* __restrict__ vrow2 = (const v8h*)(&sV[wave][2][row][0]);

  v8f accS[4] = {};       // out_s: 16 x 64
  v8f accV[3][2] = {};    // out_v: per i, 16 x 32

  // ---- term 1: out_s += [s (x) s] (16x4096) . B000 (4096x64) ----
  for (int c = 0; c < 128; ++c) {
    int u  = c >> 1;
    int vg = (c & 1) * 4;                       // v-base/8
    _Float16 su = sS[wave][row][u];
    v8h lo = srow[vg + og];
    v8h hi = srow[vg + og + 2];
    V16 A; A.h[0] = lo * su; A.h[1] = hi * su;
#pragma unroll
    for (int t = 0; t < 4; ++t) {
      const v8h* bp = (const v8h*)B000 + (((size_t)t * 128 + c) * 32 + lane) * 2;
      V16 B; B.h[0] = bp[0]; B.h[1] = bp[1];
      accS[t] = __builtin_amdgcn_wmma_f32_16x16x32_f16(
          false, A.v, false, B.v, (short)0, accS[t], false, false);
    }
  }

  // ---- term 2: out_s += [v . v gram] (16x1024) . B110 (1024x64) ----
  for (int c = 0; c < 32; ++c) {
    _Float16 a0 = sV[wave][0][row][c];
    _Float16 a1 = sV[wave][1][row][c];
    _Float16 a2 = sV[wave][2][row][c];
    v8h lo = vrow0[og]     * a0 + vrow1[og]     * a1 + vrow2[og]     * a2;
    v8h hi = vrow0[og + 2] * a0 + vrow1[og + 2] * a1 + vrow2[og + 2] * a2;
    V16 A; A.h[0] = lo; A.h[1] = hi;
#pragma unroll
    for (int t = 0; t < 4; ++t) {
      const v8h* bp = (const v8h*)B110 + (((size_t)t * 32 + c) * 32 + lane) * 2;
      V16 B; B.h[0] = bp[0]; B.h[1] = bp[1];
      accS[t] = __builtin_amdgcn_wmma_f32_16x16x32_f16(
          false, A.v, false, B.v, (short)0, accS[t], false, false);
    }
  }

  // ---- term 3: out_v_i = [s (x) v_i ; v_i (x) s] (16x4096) . BV (4096x32) ----
  for (int c = 0; c < 128; ++c) {
    V16 A[3];
    if (c < 64) {                               // features s[u] * v[v,i], u = c
      _Float16 su = sS[wave][row][c];
      A[0].h[0] = vrow0[og] * su; A[0].h[1] = vrow0[og + 2] * su;
      A[1].h[0] = vrow1[og] * su; A[1].h[1] = vrow1[og + 2] * su;
      A[2].h[0] = vrow2[og] * su; A[2].h[1] = vrow2[og + 2] * su;
    } else {                                    // features v[u,i] * s[v]
      int cc = c - 64;
      int u  = cc >> 1;
      int vg = (cc & 1) * 4;
      v8h lo = srow[vg + og];
      v8h hi = srow[vg + og + 2];
      _Float16 a0 = sV[wave][0][row][u];
      _Float16 a1 = sV[wave][1][row][u];
      _Float16 a2 = sV[wave][2][row][u];
      A[0].h[0] = lo * a0; A[0].h[1] = hi * a0;
      A[1].h[0] = lo * a1; A[1].h[1] = hi * a1;
      A[2].h[0] = lo * a2; A[2].h[1] = hi * a2;
    }
#pragma unroll
    for (int t = 0; t < 2; ++t) {
      const v8h* bp = (const v8h*)BV + (((size_t)t * 128 + c) * 32 + lane) * 2;
      V16 B; B.h[0] = bp[0]; B.h[1] = bp[1];
#pragma unroll
      for (int i = 0; i < 3; ++i) {
        accV[i][t] = __builtin_amdgcn_wmma_f32_16x16x32_f16(
            false, A[i].v, false, B.v, (short)0, accV[i][t], false, false);
      }
    }
  }

  // ---- store: C layout -> VGPR r, lane: M = (lane>>4)*8 + r, N = lane&15 ----
  float* __restrict__ orow = out + (size_t)b0 * 160;
  const int Mb = og * 8;
  const int nc = lane & 15;
#pragma unroll
  for (int t = 0; t < 4; ++t)
#pragma unroll
    for (int r = 0; r < 8; ++r)
      orow[(Mb + r) * 160 + t * 16 + nc] = accS[t][r];
#pragma unroll
  for (int i = 0; i < 3; ++i)
#pragma unroll
    for (int t = 0; t < 2; ++t)
#pragma unroll
      for (int r = 0; r < 8; ++r)
        orow[(Mb + r) * 160 + 64 + (t * 16 + nc) * 3 + i] = accV[i][t][r];
}

// -----------------------------------------------------------------------------
extern "C" void kernel_launch(void* const* d_in, const int* in_sizes, int n_in,
                              void* d_out, int out_size, void* d_ws, size_t ws_size,
                              hipStream_t stream) {
  const float* x    = (const float*)d_in[0];
  const float* W000 = (const float*)d_in[1];
  const float* W110 = (const float*)d_in[2];
  const float* W011 = (const float*)d_in[3];
  const float* W101 = (const float*)d_in[4];
  float* out = (float*)d_out;

  // workspace layout (f16 halves): B000 262144 | B110 65536 | BV 131072 = 896 KB
  _Float16* B000 = (_Float16*)d_ws;
  _Float16* B110 = B000 + 262144;
  _Float16* BV   = B110 + 65536;

  prep_B000<<<262144 / 256, 256, 0, stream>>>(W000, B000);
  prep_B110<<<65536 / 256, 256, 0, stream>>>(W110, B110);
  prep_BV  <<<131072 / 256, 256, 0, stream>>>(W011, W101, BV);

  dim3 grid((NTILES + WAVES - 1) / WAVES);   // 391 blocks x 256 threads
  si_main<<<grid, 256, 0, stream>>>(x, B000, B110, BV, out, NTILES);
}
